// Attention_62130996904205
// MI455X (gfx1250) — matrix operand order repaired and checked by reading, compiled, not run on hardware
//
#include <hip/hip_runtime.h>

typedef __attribute__((ext_vector_type(16))) _Float16 v16h;
typedef __attribute__((ext_vector_type(8)))  _Float16 v8h;
typedef __attribute__((ext_vector_type(8)))  float    v8f;

constexpr int B_  = 4;
constexpr int M_  = 1024;
constexpr int NK_ = 4096;
constexpr int NW_ = NK_ / 32;          // mask words per row = 128
constexpr int DQ_ = 512;
constexpr int DC_ = 512;
constexpr int H_  = 8;
constexpr int DH_ = 64;
constexpr int D_  = H_ * DH_;          // 512
constexpr float NEGBIG = -10000000.0f;
constexpr float SCALE  = 0.125f;       // DH^-0.5

// ---------------------------------------------------------------- helpers

__device__ __forceinline__ v8f wmma16(v16h a, v16h b, v8f c) {
  // V_WMMA_F32_16X16X32_F16 : D = A(16x32 f16) * B(32x16 f16) + C(16x16 f32)
  return __builtin_amdgcn_wmma_f32_16x16x32_f16(
      /*neg_a=*/false, a, /*neg_b=*/false, b,
      /*c_mod=*/(short)0, c, /*reuse_a=*/false, /*reuse_b=*/false);
}

__device__ __forceinline__ v16h pack_a(v8h lo, v8h hi) {
  v16h a;
#pragma unroll
  for (int i = 0; i < 8; ++i) { a[i] = lo[i]; a[i + 8] = hi[i]; }
  return a;
}

// A fragment (16x32 f16, row m = lane&15):
//   lane group 0 (lanes 0-15):  K = 0..7 (lo), 8..15 (hi)
//   lane group 1 (lanes 16-31): K = 16..23 (lo), 24..31 (hi)
__device__ __forceinline__ v16h load_a_frag(const _Float16* rowp, int kbase, int kg) {
  v8h lo = *(const v8h*)(rowp + kbase + kg * 16);
  v8h hi = *(const v8h*)(rowp + kbase + kg * 16 + 8);
  return pack_a(lo, hi);
}

// ---------------------------------------------------------------- converts

__global__ __launch_bounds__(256) void cvt_f32_to_f16_kernel(
    const float* __restrict__ src, _Float16* __restrict__ dst, int n) {
  int i = blockIdx.x * blockDim.x + threadIdx.x;
  if (i < n) dst[i] = (_Float16)src[i];
}

// src is (R x C) row-major fp32; dst is (C x R) row-major f16 (i.e. transposed)
__global__ __launch_bounds__(256) void transpose_cvt_kernel(
    const float* __restrict__ src, _Float16* __restrict__ dst, int R, int C) {
  int i = blockIdx.x * blockDim.x + threadIdx.x;
  if (i >= R * C) return;
  int c = i % C, r = i / C;
  dst[(size_t)c * R + r] = (_Float16)src[(size_t)r * C + c];
}

// kv16 is (B*NK x 2D) f16; extract V half into vT[(b,h,dh,n)]
__global__ __launch_bounds__(256) void build_vT_kernel(
    const _Float16* __restrict__ kv16, _Float16* __restrict__ vT) {
  int i = blockIdx.x * blockDim.x + threadIdx.x;
  if (i >= B_ * NK_ * D_) return;
  int d  = i % D_;
  int bn = i / D_;
  int n  = bn % NK_;
  int b  = bn / NK_;
  int h  = d / DH_, dh = d % DH_;
  vT[(((size_t)b * H_ + h) * DH_ + dh) * NK_ + n] =
      kv16[(size_t)bn * (2 * D_) + D_ + d];
}

// Bit-pack boolean mask along N: one uint32 covers 32 consecutive keys.
__global__ __launch_bounds__(256) void pack_mask_kernel(
    const unsigned char* __restrict__ mask, unsigned int* __restrict__ mbits,
    int nwords) {
  int i = blockIdx.x * blockDim.x + threadIdx.x;
  if (i >= nwords) return;
  const uint4* p = (const uint4*)(mask + (size_t)i * 32);  // 32B aligned
  uint4 a = p[0], bq = p[1];
  unsigned int w = 0;
  unsigned int words[8] = {a.x, a.y, a.z, a.w, bq.x, bq.y, bq.z, bq.w};
#pragma unroll
  for (int q = 0; q < 8; ++q)
#pragma unroll
    for (int j = 0; j < 4; ++j)
      w |= (((words[q] >> (8 * j)) & 0xffu) ? 1u : 0u) << (q * 4 + j);
  mbits[i] = w;
}

// ---------------------------------------------------------------- WMMA GEMM
// C(M x N) = A(M x K, f16 row-major) * BT(N x K, f16 row-major)^T
// One wave per 16x16 output tile; 4 waves per block.

template <bool F16OUT>
__global__ __launch_bounds__(128) void gemm_wmma_kernel(
    const _Float16* __restrict__ A, const _Float16* __restrict__ BT,
    void* __restrict__ Cout, const float* __restrict__ bias,
    int Nt, int K, int ldc) {
  int wid  = threadIdx.x >> 5;
  int lane = threadIdx.x & 31;
  int tn = blockIdx.x * 4 + wid;
  int tm = blockIdx.y;
  if (tn >= Nt) return;
  int col = lane & 15;
  int kg  = (lane >> 4) & 1;

  const _Float16* arow = A  + (size_t)(tm * 16 + col) * K;  // A: lane&15 = M
  const _Float16* brow = BT + (size_t)(tn * 16 + col) * K;  // B: lane&15 = N
  v8f acc = {};

  for (int k0 = 0; k0 < K; k0 += 32) {
    __builtin_prefetch(arow + k0 + 128, 0, 1);
    v16h a = load_a_frag(arow, k0, kg);
    v16h b = *(const v16h*)(brow + k0 + kg * 16);  // K = kg*16 + e, contiguous
    acc = wmma16(a, b, acc);
  }

  int nglob = tn * 16 + col;
#pragma unroll
  for (int e = 0; e < 8; ++e) {
    int mglob = tm * 16 + 8 * kg + e;  // C layout: m = 8*(lane>=16) + vgpr
    if (F16OUT) {
      ((_Float16*)Cout)[(size_t)mglob * ldc + nglob] = (_Float16)acc[e];
    } else {
      ((float*)Cout)[(size_t)mglob * ldc + nglob] = acc[e] + bias[nglob];
    }
  }
}

// ---------------------------------------------------------------- attention
// One wave per (b, h, 16-row M tile). Flash-style online softmax over NK
// keys, processed 32 keys per iteration (one WMMA K-step).

__global__ __launch_bounds__(32) void attn_kernel(
    const _Float16* __restrict__ q16,        // (B*M x D)
    const _Float16* __restrict__ kv16,       // (B*NK x 2D); K = cols [0,D)
    const _Float16* __restrict__ vT,         // (B,H,DH,NK)
    const unsigned int* __restrict__ mbits,  // (B,M,NK/32) packed mask
    _Float16* __restrict__ attn16) {         // (B*M x D)
  int mt = blockIdx.x, h = blockIdx.y, b = blockIdx.z;
  int lane = threadIdx.x;
  int col = lane & 15;
  int kg  = (lane >> 4) & 1;

  __shared__ __align__(32) _Float16 plds[2][16 * 32];

  // Q fragments stay resident: A-matrix rows m = lane&15, K-dim = DH (64)
  const _Float16* qrow =
      q16 + ((size_t)b * M_ + mt * 16 + col) * D_ + h * DH_;
  v16h qa0 = load_a_frag(qrow, 0, kg);
  v16h qa1 = load_a_frag(qrow, 32, kg);

  // Per-row packed-mask row pointers (row = C-fragment row for this lane).
  const unsigned int* mrowp[8];
#pragma unroll
  for (int e = 0; e < 8; ++e)
    mrowp[e] = mbits + ((size_t)b * M_ + mt * 16 + 8 * kg + e) * NW_;

  float mrow[8], lrow[8];
  v8f o0 = {}, o1 = {}, o2 = {}, o3 = {};
#pragma unroll
  for (int e = 0; e < 8; ++e) { mrow[e] = -3.0e38f; lrow[e] = 0.0f; }

  for (int n0 = 0; n0 < NK_; n0 += 32) {
    int buf = (n0 >> 5) & 1;
    // ---- S = scale * Q K^T for 16 rows x 32 keys (two 16x16 tiles) ----
    v8f s[2];
#pragma unroll
    for (int t = 0; t < 2; ++t) {
      const _Float16* krow =
          kv16 + ((size_t)b * NK_ + n0 + t * 16 + col) * (2 * D_) + h * DH_;
      v16h kb0 = *(const v16h*)(krow + kg * 16);        // dh 0..31
      v16h kb1 = *(const v16h*)(krow + 32 + kg * 16);   // dh 32..63
      v8f acc = {};
      acc = wmma16(qa0, kb0, acc);
      acc = wmma16(qa1, kb1, acc);
      s[t] = acc;
    }
    // ---- mask (packed bits) + scale ----
    unsigned int mw[8];
#pragma unroll
    for (int e = 0; e < 8; ++e) mw[e] = mrowp[e][n0 >> 5];
#pragma unroll
    for (int t = 0; t < 2; ++t)
#pragma unroll
      for (int e = 0; e < 8; ++e) {
        bool keep = (mw[e] >> (t * 16 + col)) & 1u;
        s[t][e] = keep ? s[t][e] * SCALE : NEGBIG;
      }
    // ---- online softmax: row max / rescale / exp / row sum ----
    float alpha[8];
#pragma unroll
    for (int e = 0; e < 8; ++e) {
      float v = fmaxf(s[0][e], s[1][e]);
#pragma unroll
      for (int off = 1; off < 16; off <<= 1)
        v = fmaxf(v, __shfl_xor(v, off, 32));  // reduce over 16 cols of row
      float mnew = fmaxf(mrow[e], v);
      alpha[e] = __expf(mrow[e] - mnew);
      mrow[e]  = mnew;
    }
#pragma unroll
    for (int e = 0; e < 8; ++e) {
      float p0 = __expf(s[0][e] - mrow[e]);
      float p1 = __expf(s[1][e] - mrow[e]);
      s[0][e] = p0; s[1][e] = p1;
      float rs = p0 + p1;
#pragma unroll
      for (int off = 1; off < 16; off <<= 1)
        rs += __shfl_xor(rs, off, 32);
      lrow[e] = lrow[e] * alpha[e] + rs;
      o0[e] *= alpha[e]; o1[e] *= alpha[e];
      o2[e] *= alpha[e]; o3[e] *= alpha[e];
    }
    // ---- re-layout P (C-fragment) into an A-fragment via LDS ----
#pragma unroll
    for (int t = 0; t < 2; ++t)
#pragma unroll
      for (int e = 0; e < 8; ++e)
        plds[buf][(8 * kg + e) * 32 + t * 16 + col] = (_Float16)s[t][e];
    __syncthreads();  // fence: cross-lane LDS stores -> A-fragment loads
    v8h plo = *(const v8h*)&plds[buf][col * 32 + kg * 16];
    v8h phi = *(const v8h*)&plds[buf][col * 32 + kg * 16 + 8];
    v16h pa = pack_a(plo, phi);
    // ---- O += P @ V over 4 dh-column tiles; vT rows are contiguous in n ----
    const _Float16* vbase =
        vT + (((size_t)b * H_ + h) * DH_) * NK_ + n0 + kg * 16;
    v16h vb;
    vb = *(const v16h*)(vbase + (size_t)(0 * 16 + col) * NK_); o0 = wmma16(pa, vb, o0);
    vb = *(const v16h*)(vbase + (size_t)(1 * 16 + col) * NK_); o1 = wmma16(pa, vb, o1);
    vb = *(const v16h*)(vbase + (size_t)(2 * 16 + col) * NK_); o2 = wmma16(pa, vb, o2);
    vb = *(const v16h*)(vbase + (size_t)(3 * 16 + col) * NK_); o3 = wmma16(pa, vb, o3);
  }

  // ---- epilogue: normalize by row sum, write f16 output ----
#pragma unroll
  for (int e = 0; e < 8; ++e) {
    float inv = 1.0f / lrow[e];
    int mg = mt * 16 + 8 * kg + e;
    _Float16* orow = attn16 + ((size_t)b * M_ + mg) * D_ + h * DH_;
    orow[0 * 16 + col] = (_Float16)(o0[e] * inv);
    orow[1 * 16 + col] = (_Float16)(o1[e] * inv);
    orow[2 * 16 + col] = (_Float16)(o2[e] * inv);
    orow[3 * 16 + col] = (_Float16)(o3[e] * inv);
  }
}

// ---------------------------------------------------------------- launch

extern "C" void kernel_launch(void* const* d_in, const int* in_sizes, int n_in,
                              void* d_out, int out_size, void* d_ws, size_t ws_size,
                              hipStream_t stream) {
  const float*         left  = (const float*)d_in[0];  // (B,M,DQ)
  const float*         right = (const float*)d_in[1];  // (B,NK,DC)
  const unsigned char* mask  = (const unsigned char*)d_in[2];  // (B,M,NK) bool
  const float*         Wq    = (const float*)d_in[3];  // (DQ,D)
  const float*         Wkv   = (const float*)d_in[4];  // (DC,2D)
  const float*         Wout  = (const float*)d_in[5];  // (D,DQ)
  const float*         bout  = (const float*)d_in[6];  // (DQ)

  // workspace carve-up, 256B aligned
  char* ws = (char*)d_ws;
  auto carve = [&](size_t bytes) {
    char* p = ws;
    ws += (bytes + 255) & ~(size_t)255;
    return p;
  };
  _Float16* left16  = (_Float16*)carve((size_t)B_ * M_ * DQ_ * 2);
  _Float16* right16 = (_Float16*)carve((size_t)B_ * NK_ * DC_ * 2);
  _Float16* WqT     = (_Float16*)carve((size_t)D_ * DQ_ * 2);        // (D  x DQ)
  _Float16* WkvT    = (_Float16*)carve((size_t)(2 * D_) * DC_ * 2);  // (2D x DC)
  _Float16* WoutT   = (_Float16*)carve((size_t)DQ_ * D_ * 2);        // (DQ x D)
  _Float16* q16     = (_Float16*)carve((size_t)B_ * M_ * D_ * 2);
  _Float16* kv16    = (_Float16*)carve((size_t)B_ * NK_ * 2 * D_ * 2);
  _Float16* vT      = (_Float16*)carve((size_t)B_ * H_ * DH_ * NK_ * 2);
  _Float16* attn16  = (_Float16*)carve((size_t)B_ * M_ * D_ * 2);
  unsigned int* mbits = (unsigned int*)carve((size_t)B_ * M_ * NW_ * 4);

  const int TPB = 256;
  auto blocks = [&](int n) { return (n + TPB - 1) / TPB; };

  // 1) precision conversion + weight transposes + mask packing
  {
    int n = B_ * M_ * DQ_;
    cvt_f32_to_f16_kernel<<<blocks(n), TPB, 0, stream>>>(left, left16, n);
  }
  {
    int n = B_ * NK_ * DC_;
    cvt_f32_to_f16_kernel<<<blocks(n), TPB, 0, stream>>>(right, right16, n);
  }
  transpose_cvt_kernel<<<blocks(DQ_ * D_), TPB, 0, stream>>>(Wq, WqT, DQ_, D_);
  transpose_cvt_kernel<<<blocks(DC_ * 2 * D_), TPB, 0, stream>>>(Wkv, WkvT, DC_, 2 * D_);
  transpose_cvt_kernel<<<blocks(D_ * DQ_), TPB, 0, stream>>>(Wout, WoutT, D_, DQ_);
  {
    int nwords = B_ * M_ * NW_;
    pack_mask_kernel<<<blocks(nwords), TPB, 0, stream>>>(mask, mbits, nwords);
  }

  // 2) q = left @ Wq  -> (B*M x D), f16
  {
    int Mt = (B_ * M_) / 16, Nt = D_ / 16;
    dim3 grid((Nt + 3) / 4, Mt);
    gemm_wmma_kernel<true><<<grid, 128, 0, stream>>>(
        left16, WqT, (void*)q16, nullptr, Nt, DQ_, D_);
  }
  // 3) kv = right @ Wkv -> (B*NK x 2D), f16
  {
    int Mt = (B_ * NK_) / 16, Nt = (2 * D_) / 16;
    dim3 grid((Nt + 3) / 4, Mt);
    gemm_wmma_kernel<true><<<grid, 128, 0, stream>>>(
        right16, WkvT, (void*)kv16, nullptr, Nt, DC_, 2 * D_);
  }
  // 4) V transpose for contiguous P@V B-fragment loads
  build_vT_kernel<<<blocks(B_ * NK_ * D_), TPB, 0, stream>>>(kv16, vT);

  // 5) flash attention
  {
    dim3 grid(M_ / 16, H_, B_);
    attn_kernel<<<grid, 32, 0, stream>>>(q16, kv16, vT, mbits, attn16);
  }
  // 6) out = attn @ Wout + bias -> fp32
  {
    int Mt = (B_ * M_) / 16, Nt = DQ_ / 16;
    dim3 grid((Nt + 3) / 4, Mt);
    gemm_wmma_kernel<false><<<grid, 128, 0, stream>>>(
        attn16, WoutT, d_out, bout, Nt, D_, DQ_);
  }
  (void)in_sizes; (void)n_in; (void)out_size; (void)ws_size;
}
